// MultiHeadAttention_77584289235368
// MI455X (gfx1250) — matrix operand order repaired and checked by reading, not compiled, not run on hardware
//
#include <hip/hip_runtime.h>

// ---------------------------------------------------------------------------
// MultiHeadAttention forward for MI455X (gfx1250, wave32, WMMA 16x16x32 f16).
//   x(4,2048,768) @ W_qkv(768,2304)+b -> per-head Q,K,V -> softmax(QK^T/sqrt48)V
//   -> @ W_o + b_o -> @ W_p + b_p
// f16 storage / f32 accumulation via v_wmma_f32_16x16x32_f16.
// GEMMs: TDM (tensor_load_to_lds) double-buffered tile DMA, pad feature used
// to produce the +8-half LDS row padding. Attention: flash style, S^T trick,
// global_prefetch for the streaming K/V blocks.
// ---------------------------------------------------------------------------

typedef __attribute__((ext_vector_type(16))) _Float16     v16h;
typedef __attribute__((ext_vector_type(8)))  float        v8f;
typedef __attribute__((ext_vector_type(4)))  unsigned int v4u;
typedef __attribute__((ext_vector_type(8)))  int          v8i;
typedef __attribute__((ext_vector_type(4)))  int          v4i;

#define WMMA_F16(a, b, c) \
  __builtin_amdgcn_wmma_f32_16x16x32_f16(false, (a), false, (b), (short)0, (c), false, false)

static constexpr int Dm   = 768;
static constexpr int Hh   = 16;
static constexpr int DH   = 48;
static constexpr int DHP  = 64;     // padded head dim
static constexpr int NSEQ = 2048;
static constexpr int Bb   = 4;
static constexpr int MTOT = Bb * NSEQ;   // 8192

// LDS tile: 128 rows x 32 halves, row stride 40 halves (TDM pads 4 DW/row)
static constexpr int TROW = 40;

// ---------------------------------------------------------------- converts
__global__ void cvt_f32_f16(const float* __restrict__ src, _Float16* __restrict__ dst, int n) {
  int i = blockIdx.x * blockDim.x + threadIdx.x;
  if (i < n) dst[i] = (_Float16)src[i];
}

// dst[n][k] = src[k][n]  (weights stored transposed for contiguous B-frags)
__global__ void transpose_cvt(const float* __restrict__ src, _Float16* __restrict__ dst,
                              int K, int N) {
  int i = blockIdx.x * blockDim.x + threadIdx.x;
  if (i < K * N) {
    int n = i / K, k = i - n * K;
    dst[(size_t)n * K + k] = (_Float16)src[(size_t)k * N + n];
  }
}

// zero pad columns 48..63 of Q or K ([rows][64] layout, one row per thread)
__global__ void zero_pad64(_Float16* __restrict__ buf) {
  size_t row = (size_t)blockIdx.x * blockDim.x + threadIdx.x;
  uint4 z = {0u, 0u, 0u, 0u};
  uint4* p = (uint4*)(buf + row * DHP + DH);
  p[0] = z; p[1] = z;
}

// ---------------------------------------------------------------- TDM load
// DMA a 128-row x 32-half tile (row stride K halves) from global into LDS,
// padding 4 DWORDs after every 16 DWORDs (= one 32-half row) so the LDS row
// stride becomes 40 halves. D# layout per CDNA5 ISA ch.8.
__device__ __forceinline__ unsigned lds_offset(const void* p) {
  return (unsigned)(unsigned long long)(__attribute__((address_space(3))) const void*)p;
}

__device__ __forceinline__ void tdm_load_tile(const _Float16* gsrc, unsigned lds_byte, int K) {
  unsigned long long ga = (unsigned long long)gsrc;
  v4u g0;
  g0[0] = 1u;                                       // count=1, no gather
  g0[1] = lds_byte;                                 // lds_addr
  g0[2] = (unsigned)(ga & 0xffffffffull);           // global_addr[31:0]
  g0[3] = (unsigned)((ga >> 32) & 0x1ffffffull) | (2u << 30);  // addr[56:32] | type=2

  const unsigned td1 = 1u << 20;                    // generous tensor_dim1 (no OOB)
  v8i g1;
  g1[0] = (int)((1u << 16)                          // data_size = 2 bytes
              | (1u << 20)                          // pad_enable
              | (3u << 22)                          // pad_interval: 16 DWORDs
              | (3u << 25));                        // pad_amount:   4 DWORDs
  g1[1] = (int)(((unsigned)K & 0xffffu) << 16);     // tensor_dim0[15:0]
  g1[2] = (int)((((unsigned)K >> 16) & 0xffffu) | ((td1 & 0xffffu) << 16));
  g1[3] = (int)(((td1 >> 16) & 0xffffu) | (32u << 16));   // tile_dim0 = 32
  g1[4] = (int)(128u);                              // tile_dim1 = 128, tile_dim2 = 0
  g1[5] = (int)K;                                   // tensor_dim0_stride[31:0]
  g1[6] = 0;
  g1[7] = 0;
  v4i g2 = {0, 0, 0, 0};
  v4i g3 = {0, 0, 0, 0};
  v8i g4 = {0, 0, 0, 0, 0, 0, 0, 0};
  __builtin_amdgcn_tensor_load_to_lds(g0, g1, g2, g3, g4, 0);
}

// ---------------------------------------------------------------- GEMM
// C[M,N] = A[M,K](f16) * Bt[N,K]^T(f16) + bias.  MODE 0: f32 out, 1: f16 out,
// MODE 2: QKV epilogue scatter into Q (scaled by rsqrt(48)), K, V-transposed.
template <int MODE>
__global__ void __launch_bounds__(256)
gemm_kernel(const _Float16* __restrict__ A, const _Float16* __restrict__ Bt,
            const float* __restrict__ bias,
            float* __restrict__ outF, _Float16* __restrict__ outH,
            _Float16* __restrict__ Qb, _Float16* __restrict__ Kbuf,
            _Float16* __restrict__ Vt, int M, int N, int K) {
  __shared__ _Float16 As[2][128 * TROW];
  __shared__ _Float16 Bs[2][128 * TROW];

  const int tid  = threadIdx.x;
  const int lane = tid & 31;
  const int wave = tid >> 5;
  const int wm   = wave & 1;    // 2 m-slots of 64 rows
  const int wn   = wave >> 1;   // 4 n-slots of 32 cols
  const int m0   = blockIdx.x * 128;
  const int n0   = blockIdx.y * 128;

  v8f acc[4][2];
#pragma unroll
  for (int mi = 0; mi < 4; ++mi)
#pragma unroll
    for (int ni = 0; ni < 2; ++ni)
#pragma unroll
      for (int e = 0; e < 8; ++e) acc[mi][ni][e] = 0.0f;

  const int nk = K / 32;
  if (wave == 0) {                                  // prime first tile pair
    tdm_load_tile(A  + (size_t)m0 * K, lds_offset(&As[0][0]), K);
    tdm_load_tile(Bt + (size_t)n0 * K, lds_offset(&Bs[0][0]), K);
  }

  int cur = 0;
  for (int ik = 0; ik < nk; ++ik) {
    if (wave == 0) {
      if (ik + 1 < nk) {                            // kick off next tile pair
        tdm_load_tile(A  + (size_t)m0 * K + (ik + 1) * 32, lds_offset(&As[cur ^ 1][0]), K);
        tdm_load_tile(Bt + (size_t)n0 * K + (ik + 1) * 32, lds_offset(&Bs[cur ^ 1][0]), K);
        __builtin_amdgcn_s_wait_tensorcnt(2);       // current pair complete
      } else {
        __builtin_amdgcn_s_wait_tensorcnt(0);
      }
    }
    __syncthreads();

    v16h af[4], bf[2];
#pragma unroll
    for (int mi = 0; mi < 4; ++mi) {               // A-frag: row per lane, contiguous K
      const _Float16* p = &As[cur][(wm * 64 + mi * 16 + (lane & 15)) * TROW + ((lane & 16) ? 16 : 0)];
      ((uint4*)&af[mi])[0] = *(const uint4*)p;
      ((uint4*)&af[mi])[1] = *(const uint4*)(p + 8);
    }
#pragma unroll
    for (int ni = 0; ni < 2; ++ni) {               // B-frag: col per lane, contiguous K
      const _Float16* p = &Bs[cur][(wn * 32 + ni * 16 + (lane & 15)) * TROW + ((lane & 16) ? 16 : 0)];
      ((uint4*)&bf[ni])[0] = *(const uint4*)p;
      ((uint4*)&bf[ni])[1] = *(const uint4*)(p + 8);
    }
#pragma unroll
    for (int mi = 0; mi < 4; ++mi)
#pragma unroll
      for (int ni = 0; ni < 2; ++ni)
        acc[mi][ni] = WMMA_F16(af[mi], bf[ni], acc[mi][ni]);
    __syncthreads();
    cur ^= 1;
  }

  const int rhalf = (lane >> 4) & 1;               // C-layout: lane holds rows r + 8*rhalf
#pragma unroll
  for (int mi = 0; mi < 4; ++mi) {
#pragma unroll
    for (int ni = 0; ni < 2; ++ni) {
      int ng = n0 + wn * 32 + ni * 16 + (lane & 15);
      float bv = bias[ng];
#pragma unroll
      for (int r = 0; r < 8; ++r) {
        int mg = m0 + wm * 64 + mi * 16 + rhalf * 8 + r;
        float val = acc[mi][ni][r] + bv;
        if (MODE == 0) {
          outF[(size_t)mg * N + ng] = val;
        } else if (MODE == 1) {
          outH[(size_t)mg * N + ng] = (_Float16)val;
        } else {                                   // QKV scatter
          int h = ng / 144, rem = ng - h * 144;
          int which = rem / 48, d = rem - which * 48;
          int bidx = mg >> 11, tok = mg & 2047;
          size_t qk = ((size_t)(bidx * Hh + h) * NSEQ + tok) * DHP + d;
          if (which == 0)
            Qb[qk] = (_Float16)(val * 0.14433756729740643f);   // 1/sqrt(48) folded in
          else if (which == 1)
            Kbuf[qk] = (_Float16)val;
          else
            Vt[((size_t)(bidx * Hh + h) * DH + d) * NSEQ + tok] = (_Float16)val;
        }
      }
    }
  }
}

// ---------------------------------------------------------------- attention
// One wave per 16-query tile; flash-attention over 32-key blocks.
// S^T = WMMA(A = K rows, B = Q) puts scores directly in P's A-frag layout.
__global__ void __launch_bounds__(128)
attn_kernel(const _Float16* __restrict__ Q, const _Float16* __restrict__ Kb,
            const _Float16* __restrict__ Vt, _Float16* __restrict__ att) {
  const int lane = threadIdx.x & 31;
  const int wave = threadIdx.x >> 5;
  const int bh   = blockIdx.x;
  const int bidx = bh >> 4, h = bh & 15;
  const int q0   = (blockIdx.y * 4 + wave) * 16;
  const int hi8  = (lane & 16) ? 8 : 0;

  const _Float16* Qp = Q  + (size_t)bh * NSEQ * DHP;
  const _Float16* Kp = Kb + (size_t)bh * NSEQ * DHP;
  const _Float16* Vp = Vt + (size_t)bh * DH * NSEQ;

  v16h qf[2];                                  // Q as B-frags (contiguous per lane)
#pragma unroll
  for (int kc = 0; kc < 2; ++kc) {
    const _Float16* p = Qp + (size_t)(q0 + (lane & 15)) * DHP + kc * 32 + ((lane & 16) ? 16 : 0);
    ((uint4*)&qf[kc])[0] = *(const uint4*)p;
    ((uint4*)&qf[kc])[1] = *(const uint4*)(p + 8);
  }

  v8f o[3];
#pragma unroll
  for (int f = 0; f < 3; ++f)
#pragma unroll
    for (int e = 0; e < 8; ++e) o[f][e] = 0.0f;
  float mrow = -3.0e38f, lrow = 0.0f;

  for (int kb = 0; kb < NSEQ; kb += 32) {
    if (kb + 32 < NSEQ) {                      // prefetch next key/value block
      __builtin_prefetch(Kp + (size_t)(kb + 32 + (lane & 31)) * DHP, 0, 1);
      __builtin_prefetch(Vp + (size_t)(lane & 31) * NSEQ + kb + 32, 0, 1);
    }

    v8f s0, s1;
#pragma unroll
    for (int e = 0; e < 8; ++e) { s0[e] = 0.0f; s1[e] = 0.0f; }

#pragma unroll
    for (int kc = 0; kc < 2; ++kc) {           // S^T tile: keys kb..kb+15
      v16h ka;
      const _Float16* p = Kp + (size_t)(kb + (lane & 15)) * DHP + kc * 32 + ((lane & 16) ? 16 : 0);
      ((uint4*)&ka)[0] = *(const uint4*)p;
      ((uint4*)&ka)[1] = *(const uint4*)(p + 8);
      s0 = WMMA_F16(ka, qf[kc], s0);
    }
#pragma unroll
    for (int kc = 0; kc < 2; ++kc) {           // S^T tile: keys kb+16..kb+31
      v16h ka;
      const _Float16* p = Kp + (size_t)(kb + 16 + (lane & 15)) * DHP + kc * 32 + ((lane & 16) ? 16 : 0);
      ((uint4*)&ka)[0] = *(const uint4*)p;
      ((uint4*)&ka)[1] = *(const uint4*)(p + 8);
      s1 = WMMA_F16(ka, qf[kc], s1);
    }

    // online softmax; this lane owns query (lane&15); partner lane = lane^16
    float mx = s0[0];
#pragma unroll
    for (int r = 1; r < 8; ++r) mx = fmaxf(mx, s0[r]);
#pragma unroll
    for (int r = 0; r < 8; ++r) mx = fmaxf(mx, s1[r]);
    mx = fmaxf(mx, __shfl_xor(mx, 16, 32));
    float mnew  = fmaxf(mrow, mx);
    float alpha = __expf(mrow - mnew);

    v16h pf;
    float rs = 0.0f;
#pragma unroll
    for (int r = 0; r < 8; ++r) {
      float e0 = __expf(s0[r] - mnew);
      float e1 = __expf(s1[r] - mnew);
      rs += e0 + e1;
      pf[r]     = (_Float16)e0;                // S^T C-layout == P A-frag layout
      pf[8 + r] = (_Float16)e1;
    }
    rs += __shfl_xor(rs, 16, 32);
    lrow = lrow * alpha + rs;
    mrow = mnew;

#pragma unroll
    for (int r = 0; r < 8; ++r) {              // rescale O rows (lane holds q = r+hi8)
      float sc = __shfl(alpha, r + hi8, 32);
      o[0][r] *= sc; o[1][r] *= sc; o[2][r] *= sc;
    }

#pragma unroll
    for (int f = 0; f < 3; ++f) {              // O += P * V  (V transposed: split K to match P)
      v16h vf;
      const _Float16* p = Vp + (size_t)(f * 16 + (lane & 15)) * NSEQ + kb + ((lane & 16) ? 8 : 0);
      ((uint4*)&vf)[0] = *(const uint4*)p;
      ((uint4*)&vf)[1] = *(const uint4*)(p + 16);
      o[f] = WMMA_F16(pf, vf, o[f]);
    }
  }

#pragma unroll
  for (int r = 0; r < 8; ++r) {                // divide by row sum, store f16
    float linv = 1.0f / __shfl(lrow, r + hi8, 32);
    int tok = q0 + hi8 + r;
    size_t base = (size_t)(bidx * NSEQ + tok) * Dm + h * DH;
#pragma unroll
    for (int f = 0; f < 3; ++f)
      att[base + f * 16 + (lane & 15)] = (_Float16)(o[f][r] * linv);
  }
}

// ---------------------------------------------------------------- launch
extern "C" void kernel_launch(void* const* d_in, const int* in_sizes, int n_in,
                              void* d_out, int out_size, void* d_ws, size_t ws_size,
                              hipStream_t stream) {
  (void)in_sizes; (void)n_in; (void)out_size; (void)ws_size;
  const float* x    = (const float*)d_in[0];
  const float* Wqkv = (const float*)d_in[1];
  const float* bqkv = (const float*)d_in[2];
  const float* Wo   = (const float*)d_in[3];
  const float* bo   = (const float*)d_in[4];
  const float* Wp   = (const float*)d_in[5];
  const float* bp   = (const float*)d_in[6];
  float* out = (float*)d_out;

  _Float16* ws = (_Float16*)d_ws;
  size_t off = 0;
  _Float16* xh    = ws + off; off += (size_t)MTOT * Dm;            // x in f16
  _Float16* WqkvT = ws + off; off += (size_t)(3 * Dm) * Dm;        // [2304][768]
  _Float16* WoT   = ws + off; off += (size_t)Dm * Dm;
  _Float16* WpT   = ws + off; off += (size_t)Dm * Dm;
  _Float16* Qb    = ws + off; off += (size_t)Bb * Hh * NSEQ * DHP; // [bh][n][64]
  _Float16* Kb    = ws + off; off += (size_t)Bb * Hh * NSEQ * DHP;
  _Float16* Vt    = ws + off; off += (size_t)Bb * Hh * DH * NSEQ;  // [bh][d][n]
  _Float16* att   = ws + off; off += (size_t)MTOT * Dm;
  _Float16* out1  = ws + off; off += (size_t)MTOT * Dm;

  cvt_f32_f16<<<(MTOT * Dm + 255) / 256, 256, 0, stream>>>(x, xh, MTOT * Dm);
  transpose_cvt<<<(Dm * 3 * Dm + 255) / 256, 256, 0, stream>>>(Wqkv, WqkvT, Dm, 3 * Dm);
  transpose_cvt<<<(Dm * Dm + 255) / 256, 256, 0, stream>>>(Wo, WoT, Dm, Dm);
  transpose_cvt<<<(Dm * Dm + 255) / 256, 256, 0, stream>>>(Wp, WpT, Dm, Dm);
  zero_pad64<<<(Bb * Hh * NSEQ) / 256, 256, 0, stream>>>(Qb);
  zero_pad64<<<(Bb * Hh * NSEQ) / 256, 256, 0, stream>>>(Kb);

  gemm_kernel<2><<<dim3(MTOT / 128, (3 * Dm) / 128), 256, 0, stream>>>(
      xh, WqkvT, bqkv, nullptr, nullptr, Qb, Kb, Vt, MTOT, 3 * Dm, Dm);

  attn_kernel<<<dim3(Bb * Hh, NSEQ / 64), 128, 0, stream>>>(Qb, Kb, Vt, att);

  gemm_kernel<1><<<dim3(MTOT / 128, Dm / 128), 256, 0, stream>>>(
      att, WoT, bo, nullptr, out1, nullptr, nullptr, nullptr, MTOT, Dm, Dm);
  gemm_kernel<0><<<dim3(MTOT / 128, Dm / 128), 256, 0, stream>>>(
      out1, WpT, bp, out, nullptr, nullptr, nullptr, nullptr, MTOT, Dm, Dm);
}
